// DANN_23149873725869
// MI455X (gfx1250) — matrix operand order repaired and checked by reading, not compiled, not run on hardware
//
#include <hip/hip_runtime.h>
#include <stdint.h>

typedef __attribute__((ext_vector_type(16))) _Float16 v16h;
typedef __attribute__((ext_vector_type(8)))  float    v8f;
typedef __attribute__((ext_vector_type(2)))  __fp16   fp16v2;

union HU { v16h h; uint32_t u[8]; };

// pack two f32 -> packed f16 pair in one v_cvt_pk_f16_f32
__device__ __forceinline__ uint32_t pk2h(float a, float b) {
  union { fp16v2 h; uint32_t u; } t;
  t.h = __builtin_amdgcn_cvt_pkrtz(a, b);
  return t.u;
}

// pack + ReLU in the f16 domain: v_cvt_pk_rtz_f16_f32 + v_pk_max_num_f16
__device__ __forceinline__ uint32_t pk2h_relu(float a, float b) {
  fp16v2 v = __builtin_amdgcn_cvt_pkrtz(a, b);
  fp16v2 z = {(__fp16)0.f, (__fp16)0.f};
  v = __builtin_elementwise_max(v, z);
  union { fp16v2 h; uint32_t u; } t;
  t.h = v;
  return t.u;
}

// swap data between lower/upper 16 lanes of the wave32 (SWAPX16 bitmask perm)
__device__ __forceinline__ uint32_t xor16(uint32_t x) {
  return (uint32_t)__builtin_amdgcn_ds_swizzle((int)x, 0x401f);
}

__device__ __forceinline__ v8f wmma_f16(v16h a, v16h b, v8f c) {
  return __builtin_amdgcn_wmma_f32_16x16x32_f16(false, a, false, b, (short)0, c,
                                                false, false);
}

__global__ __launch_bounds__(256, 1) void dann_fused_kernel(
    const float* __restrict__ QT_uv,
    const float* __restrict__ Vcont,
    const float* __restrict__ Vdisc,
    const float* __restrict__ Vcurv,
    const float* __restrict__ w1, const float* __restrict__ b1,
    const float* __restrict__ w2, const float* __restrict__ b2,
    const float* __restrict__ w3, const float* __restrict__ b3,
    const int* __restrict__ T,
    const int* __restrict__ QT_idx,
    const unsigned char* __restrict__ VisCont,
    const int* __restrict__ Tadj,
    const unsigned char* __restrict__ Tsign,
    const int* __restrict__ seco,
    const int* __restrict__ pTC,
    float* __restrict__ out,
    int qn, int ngroups)
{
  const int lane = (int)(threadIdx.x & 31u);
  const int col  = lane & 15;          // query column / output-feature column
  const int hi   = lane >> 4;          // half-wave id
  const int TC   = pTC[0];

  // ---- preload transposed weights as WMMA A operands (h^T = W^T @ x^T) ----
  // A layout used here (16-bit, 16x32): lane<16 holds M=lane, k = e; lane>=16
  // holds M=lane-16, k = 16+e.
  HU A1[4];                  // w1^T: 64 out-rows (4 tiles) x K=32 (8 real)
#pragma unroll
  for (int j = 0; j < 4; ++j)
#pragma unroll
    for (int e = 0; e < 16; ++e) {
      float w = (!hi && e < 8) ? w1[e * 64 + 16 * j + col] : 0.f;
      A1[j].h[e] = (_Float16)w;
    }

  HU A2[4][2];               // w2^T: 64 out-rows (4 tiles) x K=64 (2 chunks)
#pragma unroll
  for (int j = 0; j < 4; ++j)
#pragma unroll
    for (int c = 0; c < 2; ++c)
#pragma unroll
      for (int e = 0; e < 16; ++e) {
        int k = 32 * c + 16 * hi + e;
        A2[j][c].h[e] = (_Float16)w2[k * 64 + 16 * j + col];
      }

  HU A3[2];                  // w3^T: 3 out-rows (padded to 16) x K=64
#pragma unroll
  for (int c = 0; c < 2; ++c)
#pragma unroll
    for (int e = 0; e < 16; ++e) {
      int k = 32 * c + 16 * hi + e;
      float w = (col < 3) ? w3[k * 3 + col] : 0.f;
      A3[c].h[e] = (_Float16)w;
    }

  // Biases as C initializers. D layout: lane holds column n=col, rows m=8*hi+r.
  v8f C1[4], C2[4], C3;
#pragma unroll
  for (int j = 0; j < 4; ++j)
#pragma unroll
    for (int r = 0; r < 8; ++r) {
      C1[j][r] = b1[16 * j + 8 * hi + r];
      C2[j][r] = b2[16 * j + 8 * hi + r];
    }
#pragma unroll
  for (int r = 0; r < 8; ++r)
    C3[r] = (!hi && r < 3) ? b3[r] : 0.f;

  const int wid = (int)((blockIdx.x * blockDim.x + threadIdx.x) >> 5);
  const int nw  = (int)((gridDim.x * blockDim.x) >> 5);

  for (int g = wid; g < ngroups; g += nw) {
    // ---------------- gather + barycentric interpolation (1 query / lane) ---
    int q = g * 32 + lane;
    if (q >= qn) q = qn - 1;             // clamp tail; EXEC stays all-ones

    const float u  = QT_uv[2 * q + 0];
    const float v  = QT_uv[2 * q + 1];
    const float w0 = 1.f - u - v;

    const int t   = QT_idx[q];
    const int tv0 = T[3 * t + 0];
    const int tv1 = T[3 * t + 1];
    const int tv2 = T[3 * t + 2];

    int d0 = Tadj[6 * t + 0];
    int d1 = Tadj[6 * t + 2];
    int d2 = Tadj[6 * t + 4];

    const bool curved = (t < TC);
    const int  cid    = curved ? t : (TC - 1);
    const int  s0     = seco[2 * cid + 0];
    const int  s1     = seco[2 * cid + 1];
    if (curved && s0 >= 0) d1 = s0;
    if (curved && s1 >= 0) d2 = s1;

    const float* p0 = (!VisCont[tv0] && d0 >= 0) ? (Vdisc + 8 * d0) : (Vcont + 8 * tv0);
    const float* p1 = (!VisCont[tv1] && d1 >= 0) ? (Vdisc + 8 * d1) : (Vcont + 8 * tv1);
    const float* p2 = (!VisCont[tv2] && d2 >= 0) ? (Vdisc + 8 * d2) : (Vcont + 8 * tv2);
    const float* pc = Vcurv + 8 * cid;

    const float sgn = Tsign[cid] ? 1.f : -1.f;
    const float cc  = curved ? (sgn * 27.f * w0 * u * v) : 0.f;

    float feat[8];
#pragma unroll
    for (int f = 0; f < 8; ++f)
      feat[f] = w0 * p0[f] + u * p1[f] + v * p2[f] + cc * pc[f];

    uint32_t fp[4], fq[4];
#pragma unroll
    for (int i = 0; i < 4; ++i) fp[i] = pk2h(feat[2 * i], feat[2 * i + 1]);
#pragma unroll
    for (int i = 0; i < 4; ++i) fq[i] = xor16(fp[i]);

    // ---------------- MLP on WMMA, two query-tiles of 16 --------------------
#pragma unroll
    for (int qt = 0; qt < 2; ++qt) {
      // B1 = feat^T (K=32 padded from 8, N=16 queries).
      // B layout: lane<16 -> n=lane, k=0..15 in elems; lane>=16 -> k=16..31.
      HU B1;
#pragma unroll
      for (int i = 0; i < 4; ++i) {
        B1.u[i]     = hi ? 0u : (qt ? fq[i] : fp[i]);
        B1.u[4 + i] = 0u;
      }

      // Layer 1: h1^T tiles, pack+relu in f16, half-wave swap.
      uint32_t pk1[4][4], sk1[4][4];
#pragma unroll
      for (int j = 0; j < 4; ++j) {
        v8f a = wmma_f16(A1[j].h, B1.h, C1[j]);
#pragma unroll
        for (int i = 0; i < 4; ++i)
          pk1[j][i] = pk2h_relu(a[2 * i], a[2 * i + 1]);
      }
#pragma unroll
      for (int j = 0; j < 4; ++j)
#pragma unroll
        for (int i = 0; i < 4; ++i) sk1[j][i] = xor16(pk1[j][i]);

      // Layer 2: B chunks assembled from own + swapped halves (m 0..7 / 8..15)
      HU B2[2];
#pragma unroll
      for (int c = 0; c < 2; ++c)
#pragma unroll
        for (int i = 0; i < 4; ++i) {
          B2[c].u[i]     = hi ? sk1[2 * c + 1][i] : pk1[2 * c][i];
          B2[c].u[4 + i] = hi ? pk1[2 * c + 1][i] : sk1[2 * c][i];
        }

      uint32_t pk2[4][4], sk2[4][4];
#pragma unroll
      for (int j = 0; j < 4; ++j) {
        v8f a = C2[j];
        a = wmma_f16(A2[j][0].h, B2[0].h, a);
        a = wmma_f16(A2[j][1].h, B2[1].h, a);
#pragma unroll
        for (int i = 0; i < 4; ++i)
          pk2[j][i] = pk2h_relu(a[2 * i], a[2 * i + 1]);
      }
#pragma unroll
      for (int j = 0; j < 4; ++j)
#pragma unroll
        for (int i = 0; i < 4; ++i) sk2[j][i] = xor16(pk2[j][i]);

      // Layer 3 (OUT=3 padded to 16 rows)
      HU B3[2];
#pragma unroll
      for (int c = 0; c < 2; ++c)
#pragma unroll
        for (int i = 0; i < 4; ++i) {
          B3[c].u[i]     = hi ? sk2[2 * c + 1][i] : pk2[2 * c][i];
          B3[c].u[4 + i] = hi ? pk2[2 * c + 1][i] : sk2[2 * c][i];
        }
      v8f o = C3;
      o = wmma_f16(A3[0].h, B3[0].h, o);
      o = wmma_f16(A3[1].h, B3[1].h, o);

      const int oq = g * 32 + 16 * qt + col;
      if (!hi && oq < qn) {
        out[3 * oq + 0] = o[0];
        out[3 * oq + 1] = o[1];
        out[3 * oq + 2] = o[2];
      }
    }
  }
}

extern "C" void kernel_launch(void* const* d_in, const int* in_sizes, int n_in,
                              void* d_out, int out_size, void* d_ws, size_t ws_size,
                              hipStream_t stream) {
  // setup_inputs order:
  // 0 Q, 1 V (unused), 2 QT_uv, 3 V_continuous_feat, 4 V_discontinuous_feat,
  // 5 V_curved_feat, 6 w1, 7 b1, 8 w2, 9 b2, 10 w3, 11 b3, 12 T, 13 QT_idx,
  // 14 V_is_continuous(bool), 15 T_adj_disc_feat_idx, 16 T_inside_cubic_sign(bool),
  // 17 seco_T_adj_disc_feat_idx, 18 T_NUM_CURVE
  const float* QT_uv = (const float*)d_in[2];
  const float* Vcont = (const float*)d_in[3];
  const float* Vdisc = (const float*)d_in[4];
  const float* Vcurv = (const float*)d_in[5];
  const float* w1 = (const float*)d_in[6];
  const float* b1 = (const float*)d_in[7];
  const float* w2 = (const float*)d_in[8];
  const float* b2 = (const float*)d_in[9];
  const float* w3 = (const float*)d_in[10];
  const float* b3 = (const float*)d_in[11];
  const int* T = (const int*)d_in[12];
  const int* QT_idx = (const int*)d_in[13];
  const unsigned char* VisCont = (const unsigned char*)d_in[14];
  const int* Tadj = (const int*)d_in[15];
  const unsigned char* Tsign = (const unsigned char*)d_in[16];
  const int* seco = (const int*)d_in[17];
  const int* pTC = (const int*)d_in[18];
  float* out = (float*)d_out;

  const int qn = in_sizes[13];             // number of queries
  const int ngroups = (qn + 31) / 32;      // 32 queries per wave-iteration
  int blocks = (ngroups + 7) / 8;          // 8 waves per 256-thread block
  if (blocks > 1024) blocks = 1024;        // grid-stride amortizes weight preload
  if (blocks < 1) blocks = 1;

  dann_fused_kernel<<<blocks, 256, 0, stream>>>(
      QT_uv, Vcont, Vdisc, Vcurv, w1, b1, w2, b2, w3, b3,
      T, QT_idx, VisCont, Tadj, Tsign, seco, pTC, out, qn, ngroups);
}
